// VanillaLSTM_54700703482183
// MI455X (gfx1250) — compile-verified, matrix operand order checked by reading, not run on hardware
//
#include <hip/hip_runtime.h>

#define BB   32768
#define TT   128
#define INW  2
#define EMD  128
#define CELL 128
#define OUTC 5
#define EPS  1e-5f
#define ROWS 32            // batch rows per workgroup in main kernel
#define NBLK (BB / ROWS)   // 1024 workgroups in main kernel

typedef __attribute__((ext_vector_type(16))) _Float16 v16h;
typedef __attribute__((ext_vector_type(8)))  _Float16 v8h;
typedef __attribute__((ext_vector_type(8)))  float    v8f;

__device__ __forceinline__ float sigmoidf_(float x) { return 1.f / (1.f + __expf(-x)); }
__device__ __forceinline__ float tanhf_(float x) {
  float e = __expf(-2.f * x);
  return (1.f - e) / (1.f + e);
}

// Load a 16x32 f16 A-style fragment (also used for B as W rows, lane<->N):
// lane&15 selects the row; lane>>4 selects K-half offset (0 or 8) per ISA layout.
__device__ __forceinline__ v16h ldfrag(const _Float16* row, int kc, int koff) {
  v8h lo = *(const v8h*)(row + kc * 32 + koff);
  v8h hi = *(const v8h*)(row + kc * 32 + 16 + koff);
  return __builtin_shufflevector(lo, hi, 0, 1, 2, 3, 4, 5, 6, 7, 8, 9, 10, 11, 12, 13, 14, 15);
}

__device__ __forceinline__ v8f wmma_f16(v16h a, v16h b, v8f c) {
  return __builtin_amdgcn_wmma_f32_16x16x32_f16(false, a, false, b, (short)0, c,
                                                false, false);
}

// ---------------- Kernel 0: per-t batch moments of x (deterministic tree reduce) --------
__global__ __launch_bounds__(256) void xstats_kernel(const float* __restrict__ x,
                                                     float* __restrict__ part1) {
  __shared__ float red[256];
  const int t = blockIdx.x, chunk = blockIdx.y, tid = threadIdx.x;
  float s[5] = {0.f, 0.f, 0.f, 0.f, 0.f};
  for (int b = chunk * 4096 + tid; b < chunk * 4096 + 4096; b += 256) {
    float x0 = x[((size_t)b * TT + t) * INW + 0];
    float x1 = x[((size_t)b * TT + t) * INW + 1];
    s[0] += x0; s[1] += x1; s[2] += x0 * x0; s[3] += x1 * x1; s[4] += x0 * x1;
  }
  for (int i = 0; i < 5; ++i) {
    red[tid] = s[i];
    __syncthreads();
    for (int off = 128; off > 0; off >>= 1) {
      if (tid < off) red[tid] += red[tid + off];
      __syncthreads();
    }
    if (tid == 0) part1[((size_t)t * 8 + chunk) * 5 + i] = red[0];
    __syncthreads();
  }
}

// ---------------- Kernel 1: BN1 collapses to per-(t,j) affine of (x0,x1) ----------------
__global__ __launch_bounds__(256) void coef1_kernel(const float* __restrict__ part1,
                                                    const float* __restrict__ w_emb,
                                                    const float* __restrict__ gamma1,
                                                    const float* __restrict__ beta1,
                                                    float* __restrict__ coef1) {
  int idx = blockIdx.x * 256 + threadIdx.x;
  if (idx >= TT * EMD) return;
  int t = idx >> 7, j = idx & (EMD - 1);
  float S[5] = {0.f, 0.f, 0.f, 0.f, 0.f};
  for (int c = 0; c < 8; ++c)
    for (int i = 0; i < 5; ++i) S[i] += part1[((size_t)t * 8 + c) * 5 + i];
  const float invB = 1.f / (float)BB;
  float m0 = S[0] * invB, m1 = S[1] * invB;
  float v00 = S[2] * invB - m0 * m0;
  float v11 = S[3] * invB - m1 * m1;
  float v01 = S[4] * invB - m0 * m1;
  float w0 = w_emb[j * INW + 0], w1 = w_emb[j * INW + 1];
  float var = w0 * w0 * v00 + w1 * w1 * v11 + 2.f * w0 * w1 * v01;
  float rs = rsqrtf(var + EPS);
  float a0 = gamma1[j] * w0 * rs;
  float a1 = gamma1[j] * w1 * rs;
  float d = beta1[j] - a0 * m0 - a1 * m1;  // b_emb cancels in train-mode BN
  float* cf = coef1 + (size_t)idx * 3;
  cf[0] = a0; cf[1] = a1; cf[2] = d;
}

// ---------------- Kernel 2: fused persistent LSTM (WMMA f16, weights in LDS) ------------
__global__ __launch_bounds__(256) void lstm_main(
    const float* __restrict__ x, const float* __restrict__ coef1,
    const float* __restrict__ w_ih, const float* __restrict__ w_hh,
    const float* __restrict__ b_ih, const float* __restrict__ b_hh,
    const float* __restrict__ w_out, const float* __restrict__ b_out,
    float* __restrict__ ypre, float* __restrict__ part2) {
  __shared__ __align__(16) _Float16 wih_s[4 * CELL * EMD];   // 128 KB
  __shared__ __align__(16) _Float16 whh_s[4 * CELL * CELL];  // 128 KB
  __shared__ __align__(16) _Float16 e_s[ROWS * EMD];         // 8 KB
  __shared__ __align__(16) _Float16 h_s[ROWS * CELL];        // 8 KB
  __shared__ __align__(16) _Float16 wout_s[16 * CELL];       // 4 KB (rows 5..15 zero)
  __shared__ __align__(16) float    xt[ROWS * INW];          // x staging for current t
  __shared__ float ysum[64];
  __shared__ float ysq[64];

  const int tid = threadIdx.x;
  const int b0 = blockIdx.x * ROWS;

  for (int i = tid; i < 4 * CELL * EMD; i += 256)  wih_s[i] = (_Float16)w_ih[i];
  for (int i = tid; i < 4 * CELL * CELL; i += 256) whh_s[i] = (_Float16)w_hh[i];
  for (int i = tid; i < 16 * CELL; i += 256)
    wout_s[i] = (i < OUTC * CELL) ? (_Float16)w_out[i] : (_Float16)0.f;
  for (int i = tid; i < ROWS * CELL; i += 256) h_s[i] = (_Float16)0.f;
  if (tid < ROWS * INW)  // stage x for t = 0
    xt[tid] = x[(size_t)(b0 + (tid >> 1)) * (TT * INW) + (tid & 1)];

  const int lane = tid & 31;
  const int wv   = tid >> 5;            // wave 0..7 -> owns cells [wv*16, wv*16+16)
  const int mrow = lane & 15;           // A-frag row within 16-row tile
  const int koff = (lane >> 4) << 3;    // K-half per lane group (0 or 8)
  const int ncol = lane & 15;           // C/D column within tile

  float bias[4];
#pragma unroll
  for (int g = 0; g < 4; ++g) {
    int nn = g * CELL + wv * 16 + ncol;
    bias[g] = b_ih[nn] + b_hh[nn];
  }
  const float bout_lane = (ncol < OUTC) ? b_out[ncol] : 0.f;
  float cst[16];
#pragma unroll
  for (int r = 0; r < 16; ++r) cst[r] = 0.f;

  __syncthreads();

  for (int t = 0; t < TT; ++t) {
    // embedding + collapsed BN1 + ReLU -> e_s (f16); 16 channels per thread
    {
      int m = tid >> 3;            // 0..31
      int k0 = (tid & 7) * 16;     // 0..112
      float x0 = xt[m * INW + 0];
      float x1 = xt[m * INW + 1];
#pragma unroll
      for (int kk = 0; kk < 16; ++kk) {
        int k = k0 + kk;
        const float* cf = coef1 + ((size_t)t * EMD + k) * 3;
        float e = cf[0] * x0 + cf[1] * x1 + cf[2];
        e_s[m * EMD + k] = (_Float16)(e > 0.f ? e : 0.f);
      }
    }
    __syncthreads();  // B1: e_s ready (and all xt reads done)

    // prefetch x for t+1 (reads of xt for t completed before B1; next read after B4)
    if (tid < ROWS * INW && t + 1 < TT)
      xt[tid] = x[(size_t)(b0 + (tid >> 1)) * (TT * INW) + (t + 1) * INW + (tid & 1)];

    v8f acc[4][2];  // [gate i,f,g,o][m-tile]
#pragma unroll
    for (int g = 0; g < 4; ++g)
#pragma unroll
      for (int mt = 0; mt < 2; ++mt)
#pragma unroll
        for (int r = 0; r < 8; ++r) acc[g][mt][r] = bias[g];

    // kc kept rolled: 16 WMMAs per iteration with transient A/B fragments.
    // Full unroll ballooned past 256 VGPRs (s_set_vgpr_msb path) and would
    // prevent 4-waves/SIMD placement in CU mode.
#pragma unroll 1
    for (int kc = 0; kc < 4; ++kc) {
      v16h eA0 = ldfrag(e_s + mrow * EMD, kc, koff);
      v16h eA1 = ldfrag(e_s + (16 + mrow) * EMD, kc, koff);
      v16h hA0 = ldfrag(h_s + mrow * CELL, kc, koff);
      v16h hA1 = ldfrag(h_s + (16 + mrow) * CELL, kc, koff);
#pragma unroll
      for (int g = 0; g < 4; ++g) {
        const int wrow = (g * CELL + wv * 16 + ncol) * EMD;  // B-frag: weight row = N
        v16h bi = ldfrag(wih_s + wrow, kc, koff);
        acc[g][0] = wmma_f16(eA0, bi, acc[g][0]);
        acc[g][1] = wmma_f16(eA1, bi, acc[g][1]);
        v16h bh = ldfrag(whh_s + wrow, kc, koff);
        acc[g][0] = wmma_f16(hA0, bh, acc[g][0]);
        acc[g][1] = wmma_f16(hA1, bh, acc[g][1]);
      }
    }
    __syncthreads();  // B2: all h_s reads complete before overwrite

    // elementwise LSTM cell; c stays in registers, h -> LDS (f16)
#pragma unroll
    for (int mt = 0; mt < 2; ++mt) {
#pragma unroll
      for (int r = 0; r < 8; ++r) {
        float iv = sigmoidf_(acc[0][mt][r]);
        float fv = sigmoidf_(acc[1][mt][r]);
        float gv = tanhf_(acc[2][mt][r]);
        float ov = sigmoidf_(acc[3][mt][r]);
        float c = fv * cst[mt * 8 + r] + iv * gv;
        cst[mt * 8 + r] = c;
        float hv = ov * tanhf_(c);
        int m = mt * 16 + r + ((lane >> 4) << 3);
        h_s[m * CELL + wv * 16 + ncol] = (_Float16)hv;
      }
    }
    __syncthreads();  // B3: h_s ready

    // output projection via WMMA on waves 0-1 (w_out padded to 16 rows);
    // w_out B-fragments reloaded from LDS here (transient) to keep VGPRs low.
    if (wv < 2) {
      const int rowbase = wv * 16;
      v8f ay;
#pragma unroll
      for (int r = 0; r < 8; ++r) ay[r] = bout_lane;
#pragma unroll 1
      for (int kc = 0; kc < 4; ++kc) {
        v16h hA = ldfrag(h_s + (rowbase + mrow) * CELL, kc, koff);
        v16h wB = ldfrag(wout_s + ncol * CELL, kc, koff);
        ay = wmma_f16(hA, wB, ay);
      }
      float s = 0.f, sq = 0.f;
#pragma unroll
      for (int r = 0; r < 8; ++r) {
        float yv = ay[r];
        int m = rowbase + r + ((lane >> 4) << 3);
        if (ncol < OUTC) {
          ypre[((size_t)(b0 + m) * TT + t) * OUTC + ncol] = yv;
          s += yv;
          sq += yv * yv;
        }
      }
      ysum[wv * 32 + lane] = s;
      ysq[wv * 32 + lane] = sq;
    }
    __syncthreads();  // B4: ysum/ysq ready

    // deterministic fixed-order block partials for BN2 (sum, sumsq)
    if (tid < 2 * OUTC) {
      int o = tid >> 1;
      float a = (tid & 1)
                    ? (ysq[o] + ysq[o + 16] + ysq[32 + o] + ysq[32 + o + 16])
                    : (ysum[o] + ysum[o + 16] + ysum[32 + o] + ysum[32 + o + 16]);
      part2[((size_t)t * NBLK + blockIdx.x) * (2 * OUTC) + tid] = a;
    }
  }
}

// ---------------- Kernel 3: reduce BN2 partials -> per-(t,o) scale/shift ----------------
__global__ __launch_bounds__(256) void coef2_kernel(const float* __restrict__ part2,
                                                    const float* __restrict__ gamma2,
                                                    const float* __restrict__ beta2,
                                                    float* __restrict__ coef2) {
  int idx = blockIdx.x * 256 + threadIdx.x;
  if (idx >= TT * OUTC) return;
  int t = idx / OUTC, o = idx % OUTC;
  float S = 0.f, SS = 0.f;
  for (int nb = 0; nb < NBLK; ++nb) {
    const float* p = part2 + ((size_t)t * NBLK + nb) * (2 * OUTC) + o * 2;
    S += p[0];
    SS += p[1];
  }
  const float invB = 1.f / (float)BB;
  float mu = S * invB;
  float var = SS * invB - mu * mu;
  float sc = gamma2[o] * rsqrtf(var + EPS);
  coef2[t * (2 * OUTC) + o * 2 + 0] = sc;
  coef2[t * (2 * OUTC) + o * 2 + 1] = beta2[o] - mu * sc;
}

// ---------------- Kernel 4: apply BN2 + ReLU -------------------------------------------
__global__ __launch_bounds__(256) void finalize_kernel(const float* __restrict__ ypre,
                                                       const float* __restrict__ coef2,
                                                       float* __restrict__ out) {
  const size_t n = (size_t)BB * TT * OUTC;
  for (size_t i = (size_t)blockIdx.x * 256 + threadIdx.x; i < n;
       i += (size_t)gridDim.x * 256) {
    int o = (int)(i % OUTC);
    int t = (int)((i / OUTC) % TT);
    float sc = coef2[t * (2 * OUTC) + o * 2 + 0];
    float sh = coef2[t * (2 * OUTC) + o * 2 + 1];
    float y = ypre[i] * sc + sh;
    out[i] = y > 0.f ? y : 0.f;
  }
}

extern "C" void kernel_launch(void* const* d_in, const int* in_sizes, int n_in,
                              void* d_out, int out_size, void* d_ws, size_t ws_size,
                              hipStream_t stream) {
  (void)in_sizes; (void)n_in; (void)out_size; (void)ws_size;
  const float* x      = (const float*)d_in[0];
  const float* w_emb  = (const float*)d_in[1];
  // d_in[2] = b_emb (cancels inside train-mode BN1)
  const float* gamma1 = (const float*)d_in[3];
  const float* beta1  = (const float*)d_in[4];
  const float* w_ih   = (const float*)d_in[5];
  const float* w_hh   = (const float*)d_in[6];
  const float* b_ih   = (const float*)d_in[7];
  const float* b_hh   = (const float*)d_in[8];
  const float* w_out  = (const float*)d_in[9];
  const float* b_out  = (const float*)d_in[10];
  const float* gamma2 = (const float*)d_in[11];
  const float* beta2  = (const float*)d_in[12];
  float* out = (float*)d_out;

  float* ws = (float*)d_ws;
  const size_t YN = (size_t)BB * TT * OUTC;            // 21.0 M floats
  float* ypre  = ws;                                   // [B][T][5]
  float* coef1 = ypre + YN;                            // [T][128][3]
  float* part1 = coef1 + (size_t)TT * EMD * 3;         // [T][8][5]
  float* part2 = part1 + (size_t)TT * 8 * 5;           // [T][NBLK][10]
  float* coef2 = part2 + (size_t)TT * NBLK * 2 * OUTC; // [T][5][2]

  dim3 g0(TT, 8);
  xstats_kernel<<<g0, 256, 0, stream>>>(x, part1);
  coef1_kernel<<<(TT * EMD + 255) / 256, 256, 0, stream>>>(part1, w_emb, gamma1, beta1, coef1);
  lstm_main<<<NBLK, 256, 0, stream>>>(x, coef1, w_ih, w_hh, b_ih, b_hh,
                                      w_out, b_out, ypre, part2);
  coef2_kernel<<<(TT * OUTC + 255) / 256, 256, 0, stream>>>(part2, gamma2, beta2, coef2);
  finalize_kernel<<<2048, 256, 0, stream>>>(ypre, coef2, out);
}
